// Model_37357625541096
// MI455X (gfx1250) — matrix operand order repaired and checked by reading, not compiled, not run on hardware
//
#include <hip/hip_runtime.h>

// ---------------------------------------------------------------------------
// CDNA5 (gfx1250) GraphSAGE pipeline.
// GEMMs use v_wmma_f32_16x16x32_bf16 (f32 accumulate, bf16 operands staged in
// LDS).  Edge aggregation uses f32 global atomics (accumulators fit in the
// 192MB L2).  Wave32 throughout.
// ---------------------------------------------------------------------------

typedef __attribute__((ext_vector_type(16))) __bf16 v16bf;
typedef __attribute__((ext_vector_type(8)))  float  v8f;

__device__ __forceinline__ unsigned pk2bf(float lo, float hi) {
  union { float f; unsigned u; } a, b;
  a.f = lo; b.f = hi;
  unsigned rl = a.u + 0x7FFFu + ((a.u >> 16) & 1u);   // RNE f32 -> bf16
  unsigned rh = b.u + 0x7FFFu + ((b.u >> 16) & 1u);
  return (rh & 0xFFFF0000u) | (rl >> 16);
}

// ---------------------------------------------------------------------------
// WMMA GEMM:  C[m, coff+n] (ldc)  =  A[rowidx(m), :K] (lda)  @  W[K,128]
//             (+ bias[n]) (+= existing C if accum) (ReLU if relu)
// Block = 256 threads = 8 waves. Block tile 128(M) x 64(N), K-step 32.
// Wave w: wm = w&3 -> 32-row M sub-tile, wn = w>>2 -> 32-wide N strip.
// Each wave: 2 A fragments x 2 B fragments -> FOUR wmma per k-step.
// ---------------------------------------------------------------------------
#define AST 36   // As row stride (halves)
#define BST 68   // Bs row stride (halves)

__global__ void __launch_bounds__(256)
gemm_bf16_wmma(const float* __restrict__ A, int lda,
               const int* __restrict__ rowidx,
               const float* __restrict__ W,      // [K x 128] row-major
               const float* __restrict__ bias,   // [128] or nullptr
               float* __restrict__ C, int ldc, int coff,
               int M, int K, int accum, int relu)
{
  __shared__ __bf16 As[128 * AST];  // [m][k]
  __shared__ __bf16 Bs[32 * BST];   // [k][n], n = 0..63
  __shared__ int    rowA[128];

  const int tid  = threadIdx.x;
  const int lane = tid & 31;
  const int wave = tid >> 5;
  const int wm   = wave & 3;        // M offset wm*32
  const int wn   = wave >> 2;       // N strip wn*32
  const int Mbase = blockIdx.x * 128;
  const int Nbase = blockIdx.y * 64;

  if (tid < 128) {
    int gm = Mbase + tid;
    // Clamp out-of-range rows to 0: their outputs are masked in the epilogue.
    rowA[tid] = (gm < M) ? (rowidx ? rowidx[gm] : gm) : 0;
  }
  __syncthreads();

  // Precompute staging addresses (hoisted out of the k-loop).
  // A tile: 128 rows x 32 k = 1024 float4 slots, 4 per thread.
  const float* aptr[4];  int aoff[4];
  #pragma unroll
  for (int i = 0; i < 4; ++i) {
    int lin = i * 256 + tid;
    int r = lin >> 3, kv = (lin & 7) * 4;
    aptr[i] = A + (long)rowA[r] * lda + kv;
    aoff[i] = r * AST + kv;
  }
  // B tile: 32 k x 64 n = 512 float4 slots, 2 per thread.
  const float* bptr[2];  int boff[2];
  #pragma unroll
  for (int i = 0; i < 2; ++i) {
    int lin = i * 256 + tid;
    int kk = lin >> 4, nv = (lin & 15) * 4;
    bptr[i] = W + (long)kk * 128 + Nbase + nv;
    boff[i] = kk * BST + nv;
  }

  v8f acc00 = {}, acc01 = {}, acc10 = {}, acc11 = {};
  const int h     = lane >> 4;            // lane half (K-group select)
  const int arow0 = (wm * 32 + (lane & 15)) * AST;
  const int arow1 = arow0 + 16 * AST;
  const int brow  = wn * 32 + (lane & 15); // B fragment: N == lane&15

  for (int kt = 0; kt < K; kt += 32) {
    // Unconditional vector loads (no divergent guards).
    float4 a0 = *(const float4*)(aptr[0] + kt);
    float4 a1 = *(const float4*)(aptr[1] + kt);
    float4 a2 = *(const float4*)(aptr[2] + kt);
    float4 a3 = *(const float4*)(aptr[3] + kt);
    float4 b0 = *(const float4*)(bptr[0] + (long)kt * 128);
    float4 b1 = *(const float4*)(bptr[1] + (long)kt * 128);
    if (kt + 32 < K) {                    // stream-prefetch next A tile (WGP)
      __builtin_prefetch(aptr[0] + kt + 32, 0, 3);
      __builtin_prefetch(aptr[2] + kt + 32, 0, 3);
    }
    __syncthreads();                      // prior fragment reads done
    *(unsigned*)&As[aoff[0]]     = pk2bf(a0.x, a0.y);
    *(unsigned*)&As[aoff[0] + 2] = pk2bf(a0.z, a0.w);
    *(unsigned*)&As[aoff[1]]     = pk2bf(a1.x, a1.y);
    *(unsigned*)&As[aoff[1] + 2] = pk2bf(a1.z, a1.w);
    *(unsigned*)&As[aoff[2]]     = pk2bf(a2.x, a2.y);
    *(unsigned*)&As[aoff[2] + 2] = pk2bf(a2.z, a2.w);
    *(unsigned*)&As[aoff[3]]     = pk2bf(a3.x, a3.y);
    *(unsigned*)&As[aoff[3] + 2] = pk2bf(a3.z, a3.w);
    *(unsigned*)&Bs[boff[0]]     = pk2bf(b0.x, b0.y);
    *(unsigned*)&Bs[boff[0] + 2] = pk2bf(b0.z, b0.w);
    *(unsigned*)&Bs[boff[1]]     = pk2bf(b1.x, b1.y);
    *(unsigned*)&Bs[boff[1] + 2] = pk2bf(b1.z, b1.w);
    __syncthreads();

    // A fragments (16-bit A 16x32 layout, ISA 7.12.2):
    //   half h=0: vgpr0..3 -> K 0..7, vgpr4..7 -> K 16..23; h=1: +8.
    v16bf av0, av1, bv0, bv1;
    #pragma unroll
    for (int j = 0; j < 16; ++j) {
      int vg = j >> 1;
      int kk = (vg < 4 ? vg * 2 : 16 + (vg - 4) * 2) + (j & 1) + h * 8;
      av0[j] = As[arow0 + kk];
      av1[j] = As[arow1 + kk];
    }
    // B fragments (32x16 each): N == lane&15; K in same order as A.
    #pragma unroll
    for (int j = 0; j < 16; ++j) {
      int vg = j >> 1;
      int kk = (vg < 4 ? vg * 2 : 16 + (vg - 4) * 2) + (j & 1) + h * 8;
      bv0[j] = Bs[kk * BST + brow];
      bv1[j] = Bs[kk * BST + brow + 16];
    }

    acc00 = __builtin_amdgcn_wmma_f32_16x16x32_bf16(false, av0, false, bv0,
                                                    (short)0, acc00, false, false);
    acc01 = __builtin_amdgcn_wmma_f32_16x16x32_bf16(false, av0, false, bv1,
                                                    (short)0, acc01, false, false);
    acc10 = __builtin_amdgcn_wmma_f32_16x16x32_bf16(false, av1, false, bv0,
                                                    (short)0, acc10, false, false);
    acc11 = __builtin_amdgcn_wmma_f32_16x16x32_bf16(false, av1, false, bv1,
                                                    (short)0, acc11, false, false);
  }

  // Epilogue: C/D layout -> M = vgpr + h*8, N = lane&15.
  const int n0 = Nbase + wn * 32 + (lane & 15);
  const float bb0 = bias ? bias[n0]      : 0.0f;
  const float bb1 = bias ? bias[n0 + 16] : 0.0f;
  #pragma unroll
  for (int r = 0; r < 8; ++r) {
    int m0 = Mbase + wm * 32 + r + h * 8;
    int m1 = m0 + 16;
    if (m0 < M) {
      long addr = (long)m0 * ldc + coff + n0;
      float v0 = acc00[r] + bb0;
      float v1 = acc01[r] + bb1;
      if (accum) { v0 += C[addr]; v1 += C[addr + 16]; }
      if (relu)  { v0 = fmaxf(v0, 0.0f); v1 = fmaxf(v1, 0.0f); }
      C[addr]      = v0;
      C[addr + 16] = v1;
    }
    if (m1 < M) {
      long addr = (long)m1 * ldc + coff + n0;
      float v0 = acc10[r] + bb0;
      float v1 = acc11[r] + bb1;
      if (accum) { v0 += C[addr]; v1 += C[addr + 16]; }
      if (relu)  { v0 = fmaxf(v0, 0.0f); v1 = fmaxf(v1, 0.0f); }
      C[addr]      = v0;
      C[addr + 16] = v1;
    }
  }
}

// ---------------------------------------------------------------------------
// Graph / elementwise kernels
// ---------------------------------------------------------------------------
__global__ void concat_kernel(const float* __restrict__ emb,
                              const float* __restrict__ x,
                              float* __restrict__ out, int n)
{
  long idx = (long)blockIdx.x * blockDim.x + threadIdx.x;
  if (idx >= (long)n * 256) return;
  int i = (int)(idx >> 8);
  int j = (int)(idx & 255);
  out[idx] = (j < 128) ? emb[(long)i * 128 + j] : x[(long)i * 128 + j - 128];
}

__global__ void degree_kernel(const int* __restrict__ esrc,
                              const int* __restrict__ edst,
                              float* __restrict__ dU, float* __restrict__ dM,
                              int E)
{
  int e = blockIdx.x * blockDim.x + threadIdx.x;
  if (e >= E) return;
  atomicAdd(&dM[edst[e]], 1.0f);
  atomicAdd(&dU[esrc[e]], 1.0f);
}

__global__ void scatter_add4(const float* __restrict__ feat,
                             const int* __restrict__ sidx,
                             const int* __restrict__ didx,
                             float* __restrict__ agg, int E, int d)
{
  long idx = (long)blockIdx.x * blockDim.x + threadIdx.x;
  int vec = d >> 2;
  long total = (long)E * vec;
  if (idx >= total) return;
  int e = (int)(idx / vec);
  int f = (int)(idx % vec) * 4;
  int s = sidx[e], dn = didx[e];
  const float4 v = *(const float4*)(feat + (long)s * d + f);
  float* o = agg + (long)dn * d + f;
  atomicAdd(o + 0, v.x);
  atomicAdd(o + 1, v.y);
  atomicAdd(o + 2, v.z);
  atomicAdd(o + 3, v.w);
}

__global__ void mean_kernel(float* __restrict__ agg,
                            const float* __restrict__ deg, int n, int d)
{
  long idx = (long)blockIdx.x * blockDim.x + threadIdx.x;
  if (idx >= (long)n * d) return;
  int row = (int)(idx / d);
  agg[idx] = agg[idx] / fmaxf(deg[row], 1.0f);
}

__global__ void decoder_out(const float* __restrict__ z1,
                            const float* __restrict__ W2,
                            const float* __restrict__ b2,
                            float* __restrict__ out, int L)
{
  int i = blockIdx.x * blockDim.x + threadIdx.x;
  if (i >= L) return;
  const float4* z = (const float4*)(z1 + (long)i * 128);
  const float4* w = (const float4*)W2;
  float acc = 0.0f;
  #pragma unroll
  for (int j = 0; j < 32; ++j) {
    float4 a = z[j], b = w[j];
    acc += a.x * b.x + a.y * b.y + a.z * b.z + a.w * b.w;
  }
  out[i] = acc + b2[0];
}

// ---------------------------------------------------------------------------
// Host orchestration
// ---------------------------------------------------------------------------
extern "C" void kernel_launch(void* const* d_in, const int* in_sizes, int n_in,
                              void* d_out, int out_size, void* d_ws, size_t ws_size,
                              hipStream_t stream)
{
  const float* x_user    = (const float*)d_in[0];
  const float* x_movie   = (const float*)d_in[1];
  const int*   edge_src  = (const int*)d_in[2];
  const int*   edge_dst  = (const int*)d_in[3];
  const int*   label_src = (const int*)d_in[4];
  const int*   label_dst = (const int*)d_in[5];
  const float* user_emb  = (const float*)d_in[6];
  const float* movie_emb = (const float*)d_in[7];

  const int NU = in_sizes[0] / 128;
  const int NM = in_sizes[1] / 128;
  const int E  = in_sizes[2];
  const int L  = in_sizes[4];

  const float *Wn_u2m[3], *bn_u2m[3], *Wr_m[3], *Wn_m2u[3], *bn_m2u[3], *Wr_u[3];
  for (int i = 0; i < 3; ++i) {
    int b = 8 + i * 6;
    Wn_u2m[i] = (const float*)d_in[b + 0];
    bn_u2m[i] = (const float*)d_in[b + 1];
    Wr_m[i]   = (const float*)d_in[b + 2];
    Wn_m2u[i] = (const float*)d_in[b + 3];
    bn_m2u[i] = (const float*)d_in[b + 4];
    Wr_u[i]   = (const float*)d_in[b + 5];
  }
  const float* W_ui = (const float*)d_in[26];
  const float* b_ui = (const float*)d_in[27];
  const float* W_iu = (const float*)d_in[28];
  const float* b_iu = (const float*)d_in[29];
  const float* W1   = (const float*)d_in[30];
  const float* b1   = (const float*)d_in[31];
  const float* W2   = (const float*)d_in[32];
  const float* b2   = (const float*)d_in[33];
  float* out = (float*)d_out;
  (void)n_in; (void)out_size;

  // Workspace carve-up (256B aligned).
  char* p = (char*)d_ws;
  auto alloc = [&](size_t bytes) -> float* {
    float* r = (float*)p;
    p += (bytes + 255) & ~(size_t)255;
    return r;
  };
  float* hu0  = alloc((size_t)NU * 256 * 4);
  float* hm0  = alloc((size_t)NM * 256 * 4);
  float* huA  = alloc((size_t)NU * 128 * 4);
  float* huB  = alloc((size_t)NU * 128 * 4);
  float* hmA  = alloc((size_t)NM * 128 * 4);
  float* hmB  = alloc((size_t)NM * 128 * 4);
  float* aggU = alloc((size_t)NU * 256 * 4);
  float* aggM = alloc((size_t)NM * 256 * 4);
  float* degU = alloc((size_t)NU * 4);
  float* degM = alloc((size_t)NM * 4);
  float* zcat = alloc((size_t)L * 256 * 4);
  float* z1   = alloc((size_t)L * 128 * 4);
  if ((size_t)(p - (char*)d_ws) > ws_size) return;   // scratch too small

  auto gemm = [&](const float* A, int lda, const int* ridx, const float* W,
                  const float* bias, float* C, int ldc, int coff,
                  int M, int K, int accum, int relu) {
    dim3 g((unsigned)((M + 127) / 128), 2);  // 128x64 block tiles, N=128
    gemm_bf16_wmma<<<g, 256, 0, stream>>>(A, lda, ridx, W, bias, C, ldc, coff,
                                          M, K, accum, relu);
  };

  // Degrees (layer-invariant) + input concats.
  hipMemsetAsync(degU, 0, (size_t)NU * 4, stream);
  hipMemsetAsync(degM, 0, (size_t)NM * 4, stream);
  degree_kernel<<<(E + 255) / 256, 256, 0, stream>>>(edge_src, edge_dst, degU, degM, E);
  concat_kernel<<<NU, 256, 0, stream>>>(user_emb,  x_user,  hu0, NU);
  concat_kernel<<<NM, 256, 0, stream>>>(movie_emb, x_movie, hm0, NM);

  const float* inU = hu0;
  const float* inM = hm0;
  for (int i = 0; i < 3; ++i) {
    int d = (i == 0) ? 256 : 128;
    float* outU = (i == 1) ? huB : huA;
    float* outM = (i == 1) ? hmB : hmA;

    hipMemsetAsync(aggM, 0, (size_t)NM * d * 4, stream);
    hipMemsetAsync(aggU, 0, (size_t)NU * d * 4, stream);

    long t = (long)E * (d / 4);
    unsigned nb = (unsigned)((t + 255) / 256);
    scatter_add4<<<nb, 256, 0, stream>>>(inU, edge_src, edge_dst, aggM, E, d);
    scatter_add4<<<nb, 256, 0, stream>>>(inM, edge_dst, edge_src, aggU, E, d);

    mean_kernel<<<(unsigned)(((long)NM * d + 255) / 256), 256, 0, stream>>>(aggM, degM, NM, d);
    mean_kernel<<<(unsigned)(((long)NU * d + 255) / 256), 256, 0, stream>>>(aggU, degU, NU, d);

    int relu = (i < 2) ? 1 : 0;
    gemm(aggM, d, nullptr, Wn_u2m[i], bn_u2m[i], outM, 128, 0, NM, d, 0, 0);
    gemm(inM,  d, nullptr, Wr_m[i],   nullptr,   outM, 128, 0, NM, d, 1, relu);
    gemm(aggU, d, nullptr, Wn_m2u[i], bn_m2u[i], outU, 128, 0, NU, d, 0, 0);
    gemm(inU,  d, nullptr, Wr_u[i],   nullptr,   outU, 128, 0, NU, d, 1, relu);

    inU = outU;
    inM = outM;
  }

  // Edge decoder: gather fused into GEMM A-load via rowidx.
  gemm(inU, 128, label_src, W_ui, b_ui, zcat, 256, 0,   L, 128, 0, 0);
  gemm(inM, 128, label_dst, W_iu, b_iu, zcat, 256, 128, L, 128, 0, 0);
  gemm(zcat, 256, nullptr,  W1,   b1,   z1,   128, 0,   L, 256, 0, 1);
  decoder_out<<<(L + 255) / 256, 256, 0, stream>>>(z1, W2, b2, out, L);
}
